// EncoderLayer_86148454023319
// MI455X (gfx1250) — compile-verified, matrix operand order checked by reading, not run on hardware
//
#include <hip/hip_runtime.h>
#include <stdint.h>

typedef unsigned short u16;
typedef __attribute__((ext_vector_type(8)))  u16    u16x8;
typedef __attribute__((ext_vector_type(16))) u16    u16x16;
typedef __attribute__((ext_vector_type(16))) __bf16 v16bf;
typedef __attribute__((ext_vector_type(8)))  float  v8f;

#define SEQ   4096
#define DIM   768
#define NHEAD 12
#define DHEAD 64
#define WIN   32

__device__ __forceinline__ u16 f2b(float f) {
  unsigned u = __float_as_uint(f);
  unsigned r = u + 0x7FFFu + ((u >> 16) & 1u);   // round-to-nearest-even
  return (u16)(r >> 16);
}

// ---- gfx1250 async global->LDS copy (ASYNCcnt path), 16B per lane ----------
// LDS address in one VGPR, global address in a 64-bit VGPR pair (saddr off).
__device__ __forceinline__ void async_cp16(void* lds_dst, const void* gsrc) {
  unsigned d = (unsigned)(uintptr_t)(__attribute__((address_space(3))) void*)lds_dst;
  asm volatile("global_load_async_to_lds_b128 %0, %1, off"
               :: "v"(d), "v"(gsrc) : "memory");
}
__device__ __forceinline__ void wait_async0() {
  asm volatile("s_wait_asynccnt 0x0" ::: "memory");
}
__device__ __forceinline__ void wait_ds0() {
  asm volatile("s_wait_dscnt 0x0" ::: "memory");
}

// Build a 16x32 bf16 WMMA fragment (A or N-major B) from LDS.
// ISA 7.12.2: lanes 0-15 row=lane, VGPR0-3 = K 0..7, VGPR4-7 = K 16..23;
// lanes 16-31 same rows, +8 K offset.  -> two 16B loads at +0/+16 elems.
__device__ __forceinline__ v16bf ld_frag(const u16* p) {
  u16x8 lo = *(const u16x8*)p;
  u16x8 hi = *(const u16x8*)(p + 16);
  u16x16 t;
#pragma unroll
  for (int j = 0; j < 8; ++j) { t[j] = lo[j]; t[j + 8] = hi[j]; }
  return __builtin_bit_cast(v16bf, t);
}

// ---------------------------------------------------------------- pack kernels
__global__ __launch_bounds__(256)
void pack_bf16(const float* __restrict__ in, u16* __restrict__ out, int n) {
  int i = blockIdx.x * blockDim.x + threadIdx.x;
  if (i < n) out[i] = f2b(in[i]);
}

// W[k][n] fp32 -> Wt[n][k] bf16  (N-major so B-fragments are contiguous)
__global__ __launch_bounds__(256)
void pack_wT(const float* __restrict__ w, u16* __restrict__ wt) {
  int i = blockIdx.x * blockDim.x + threadIdx.x;
  if (i >= DIM * DIM) return;
  int n = i / DIM, k = i - n * DIM;
  wt[n * DIM + k] = f2b(w[k * DIM + n]);
}

// ---------------------------------------------------------------- WMMA GEMM
// C[4096,768] = (A bf16 @ Bt bf16^T + bias) * scale (opt relu)
// 128x128 block tile, 8 waves, each wave 32x64 (2x4 WMMA tiles), K-step 32.
// Async double-buffered LDS staging (GLOBAL_LOAD_ASYNC_TO_LDS_B128).
__global__ __launch_bounds__(256)
void gemm_bf16(const u16* __restrict__ A, const u16* __restrict__ Bt,
               const float* __restrict__ bias, float scale, int do_relu,
               float* __restrict__ outF, u16* __restrict__ outB)
{
  constexpr int STR = 40;                      // padded row stride (bf16 elems)
  __shared__ u16 sA[2][128 * STR];
  __shared__ u16 sB[2][128 * STR];
  const int tid  = threadIdx.x;
  const int lane = tid & 31;
  const int wave = tid >> 5;
  const int wr   = wave & 3;                   // wave row (32 rows each)
  const int wc   = wave >> 2;                  // wave col (64 cols each)
  const int rowBase = blockIdx.y * 128;
  const int colBase = blockIdx.x * 128;
  const int half = lane >> 4;
  const int l16  = lane & 15;

  // staging coordinates: 512 16B-chunks per 128x32 slab, 2 per thread
  const int r0 = tid >> 2,         q0 = (tid & 3) * 8;
  const int r1 = (tid + 256) >> 2, q1 = (tid & 3) * 8;  // same q, rows +64
  const u16* gA0 = A  + (size_t)(rowBase + r0) * DIM + q0;
  const u16* gA1 = A  + (size_t)(rowBase + r1) * DIM + q1;
  const u16* gB0 = Bt + (size_t)(colBase + r0) * DIM + q0;
  const u16* gB1 = Bt + (size_t)(colBase + r1) * DIM + q1;
  const int o0 = r0 * STR + q0;
  const int o1 = r1 * STR + q1;

  // prologue: async-stage K-slab 0 into buffer 0
  async_cp16(&sA[0][o0], gA0);
  async_cp16(&sA[0][o1], gA1);
  async_cp16(&sB[0][o0], gB0);
  async_cp16(&sB[0][o1], gB1);
  wait_async0();
  __syncthreads();

  v8f acc[2][4] = {};
  int buf = 0;
  for (int k0 = 0; k0 < DIM; k0 += 32) {
    // kick off async staging of the next K-slab into the other buffer
    if (k0 + 32 < DIM) {
      const int nb = buf ^ 1;
      async_cp16(&sA[nb][o0], gA0 + k0 + 32);
      async_cp16(&sA[nb][o1], gA1 + k0 + 32);
      async_cp16(&sB[nb][o0], gB0 + k0 + 32);
      async_cp16(&sB[nb][o1], gB1 + k0 + 32);
    }

    v16bf af[2], bfr[4];
#pragma unroll
    for (int i = 0; i < 2; ++i)
      af[i] = ld_frag(&sA[buf][(wr * 32 + i * 16 + l16) * STR + half * 8]);
#pragma unroll
    for (int j = 0; j < 4; ++j)
      bfr[j] = ld_frag(&sB[buf][(wc * 64 + j * 16 + l16) * STR + half * 8]);
#pragma unroll
    for (int i = 0; i < 2; ++i)
#pragma unroll
      for (int j = 0; j < 4; ++j)
        acc[i][j] = __builtin_amdgcn_wmma_f32_16x16x32_bf16(
            false, af[i], false, bfr[j], (short)0, acc[i][j], false, false);

    wait_async0();
    __syncthreads();
    buf ^= 1;
  }

  // epilogue: C-tile layout VGPR g -> M = g + 8*half, N = l16
#pragma unroll
  for (int i = 0; i < 2; ++i) {
#pragma unroll
    for (int j = 0; j < 4; ++j) {
      int col = colBase + wc * 64 + j * 16 + l16;
      float bv = bias ? bias[col] : 0.0f;
#pragma unroll
      for (int g = 0; g < 8; ++g) {
        int row = rowBase + wr * 32 + i * 16 + half * 8 + g;
        float v = (acc[i][j][g] + bv) * scale;
        if (do_relu) v = fmaxf(v, 0.0f);
        size_t o = (size_t)row * DIM + col;
        if (outF) outF[o] = v;
        if (outB) outB[o] = f2b(v);
      }
    }
  }
}

// ---------------------------------------------------------------- attention
// One wave per (16-query block, head).  Key slab = [qbase-32, qbase+47].
__global__ __launch_bounds__(32)
void attn_swa(const u16* __restrict__ qb, const u16* __restrict__ kb,
              const u16* __restrict__ vb, u16* __restrict__ attnb,
              float* __restrict__ probs)
{
  __shared__ u16 sQ[16 * 64];     // queries, row-major (A fragments)
  __shared__ u16 sK[80 * 64];     // keys, key-major  (B fragments, N=key)
  __shared__ u16 sV[64 * 96];     // V transposed [d][key] (B fragments, N=d)
  __shared__ u16 sP[16 * 96];     // probs bf16, zero-padded K 80..95
  const int lane  = threadIdx.x;
  const int qbase = blockIdx.x * 16;
  const int h     = blockIdx.y;
  const int kbase = qbase - WIN;
  const int half  = lane >> 4;
  const int l16   = lane & 15;

  // zero sK (so EXEC-masked async lanes leave zeros) and sP padding
  {
    u16x8 z = {};
    for (int c = lane; c < 640; c += 32) *(u16x8*)&sK[c * 8] = z;
    for (int c = lane; c < 192; c += 32) *(u16x8*)&sP[c * 8] = z;
  }
  // V transposed fill (scalar gather/transpose), zeros in the pad columns
  for (int e = lane; e < 64 * 96; e += 32) {
    int dd = e / 96, kk = e - dd * 96;
    int key = kbase + kk;
    u16 v = 0;
    if (kk < 80 && key >= 0 && key < SEQ)
      v = vb[(size_t)key * DIM + h * DHEAD + dd];
    sV[dd * 96 + kk] = v;
  }
  wait_ds0();   // zero-fill visible before async writes land

  // async-stage Q (always valid) and in-range K rows
  for (int c = lane; c < 128; c += 32) {
    int r = c >> 3, cc = c & 7;
    async_cp16(&sQ[r * 64 + cc * 8],
               &qb[(size_t)(qbase + r) * DIM + h * DHEAD + cc * 8]);
  }
  for (int c = lane; c < 640; c += 32) {
    int r = c >> 3, cc = c & 7;
    int key = kbase + r;
    if (key >= 0 && key < SEQ)
      async_cp16(&sK[r * 64 + cc * 8],
                 &kb[(size_t)key * DIM + h * DHEAD + cc * 8]);
  }
  wait_async0();
  __syncthreads();

  // scores = Qs @ K^T : M=16, N=80 (5 tiles), Kdim=64 (2 steps)
  v8f sc[5] = {};
#pragma unroll
  for (int ks = 0; ks < 2; ++ks) {
    v16bf aq = ld_frag(&sQ[l16 * 64 + ks * 32 + half * 8]);
#pragma unroll
    for (int nt = 0; nt < 5; ++nt) {
      v16bf bk = ld_frag(&sK[(nt * 16 + l16) * 64 + ks * 32 + half * 8]);
      sc[nt] = __builtin_amdgcn_wmma_f32_16x16x32_bf16(
          false, aq, false, bk, (short)0, sc[nt], false, false);
    }
  }

  // band + bounds mask, rowwise softmax (rows live in 16-lane halves)
  float mx[8], sm[8];
#pragma unroll
  for (int g = 0; g < 8; ++g) mx[g] = -3.0e38f;
#pragma unroll
  for (int nt = 0; nt < 5; ++nt)
#pragma unroll
    for (int g = 0; g < 8; ++g) {
      int r = half * 8 + g, col = nt * 16 + l16;
      int w = col - r, key = kbase + col;
      float v = (w >= 0 && w <= 2 * WIN && key >= 0 && key < SEQ)
                    ? sc[nt][g] : -1.0e9f;
      sc[nt][g] = v;
      mx[g] = fmaxf(mx[g], v);
    }
#pragma unroll
  for (int off = 8; off >= 1; off >>= 1)
#pragma unroll
    for (int g = 0; g < 8; ++g) mx[g] = fmaxf(mx[g], __shfl_xor(mx[g], off, 32));
#pragma unroll
  for (int g = 0; g < 8; ++g) sm[g] = 0.0f;
#pragma unroll
  for (int nt = 0; nt < 5; ++nt)
#pragma unroll
    for (int g = 0; g < 8; ++g) {
      float p = __expf(sc[nt][g] - mx[g]);
      sc[nt][g] = p;
      sm[g] += p;
    }
#pragma unroll
  for (int off = 8; off >= 1; off >>= 1)
#pragma unroll
    for (int g = 0; g < 8; ++g) sm[g] += __shfl_xor(sm[g], off, 32);

#pragma unroll
  for (int nt = 0; nt < 5; ++nt)
#pragma unroll
    for (int g = 0; g < 8; ++g) {
      int r = half * 8 + g, col = nt * 16 + l16;
      float p = sc[nt][g] / sm[g];
      int w = col - r;
      if (w >= 0 && w <= 2 * WIN)
        probs[((size_t)h * SEQ + (qbase + r)) * (2 * WIN + 1) + w] = p;
      sP[r * 96 + col] = f2b(p);
    }
  __syncthreads();

  // ctx = P @ V : M=16, N=64 (4 tiles), Kdim=96 (3 steps, zero-padded)
  v8f ct[4] = {};
#pragma unroll
  for (int ks = 0; ks < 3; ++ks) {
    v16bf ap = ld_frag(&sP[l16 * 96 + ks * 32 + half * 8]);
#pragma unroll
    for (int nt = 0; nt < 4; ++nt) {
      v16bf bv2 = ld_frag(&sV[(nt * 16 + l16) * 96 + ks * 32 + half * 8]);
      ct[nt] = __builtin_amdgcn_wmma_f32_16x16x32_bf16(
          false, ap, false, bv2, (short)0, ct[nt], false, false);
    }
  }
#pragma unroll
  for (int nt = 0; nt < 4; ++nt)
#pragma unroll
    for (int g = 0; g < 8; ++g) {
      int r = half * 8 + g, dd = nt * 16 + l16;
      attnb[(size_t)(qbase + r) * DIM + h * DHEAD + dd] = f2b(ct[nt][g]);
    }
}

// ---------------------------------------------------------------- layernorm
// out = LN(y + res) * g + b ; fp32 out (+ optional bf16 copy for next GEMM)
__global__ __launch_bounds__(256)
void ln_fused(const float* __restrict__ y, const float* __restrict__ res,
              const float* __restrict__ gam, const float* __restrict__ bet,
              float* __restrict__ outF, u16* __restrict__ outB)
{
  const int row = blockIdx.x;
  const int tid = threadIdx.x;
  const int lane = tid & 31, wave = tid >> 5;
  __shared__ float rs[8], rq[8];
  const float* yr = y + (size_t)row * DIM;
  const float* rr = res + (size_t)row * DIM;
  float x[3], sum = 0.f, sq = 0.f;
#pragma unroll
  for (int i = 0; i < 3; ++i) {
    int c = tid + i * 256;
    x[i] = yr[c] + rr[c];
    sum += x[i];
    sq  += x[i] * x[i];
  }
#pragma unroll
  for (int off = 16; off >= 1; off >>= 1) {
    sum += __shfl_xor(sum, off, 32);
    sq  += __shfl_xor(sq,  off, 32);
  }
  if (lane == 0) { rs[wave] = sum; rq[wave] = sq; }
  __syncthreads();
  sum = 0.f; sq = 0.f;
#pragma unroll
  for (int wv = 0; wv < 8; ++wv) { sum += rs[wv]; sq += rq[wv]; }
  float mu   = sum * (1.0f / DIM);
  float var  = sq * (1.0f / DIM) - mu * mu;
  float rstd = rsqrtf(var + 1e-5f);
#pragma unroll
  for (int i = 0; i < 3; ++i) {
    int c = tid + i * 256;
    float o = (x[i] - mu) * rstd * gam[c] + bet[c];
    outF[(size_t)row * DIM + c] = o;
    if (outB) outB[(size_t)row * DIM + c] = f2b(o);
  }
}

// ---------------------------------------------------------------- launch
extern "C" void kernel_launch(void* const* d_in, const int* in_sizes, int n_in,
                              void* d_out, int out_size, void* d_ws, size_t ws_size,
                              hipStream_t stream) {
  (void)in_sizes; (void)n_in; (void)out_size; (void)ws_size;
  const float* x   = (const float*)d_in[0];
  const float* Wq  = (const float*)d_in[1];
  const float* bq  = (const float*)d_in[2];
  const float* Wk  = (const float*)d_in[3];
  const float* bk  = (const float*)d_in[4];
  const float* Wv  = (const float*)d_in[5];
  const float* bv  = (const float*)d_in[6];
  const float* Wfc = (const float*)d_in[7];
  const float* bfc = (const float*)d_in[8];
  const float* g1  = (const float*)d_in[9];
  const float* b1  = (const float*)d_in[10];
  const float* W1  = (const float*)d_in[11];
  const float* bw1 = (const float*)d_in[12];
  const float* W2  = (const float*)d_in[13];
  const float* bw2 = (const float*)d_in[14];
  const float* g2  = (const float*)d_in[15];
  const float* b2  = (const float*)d_in[16];

  float* out   = (float*)d_out;
  float* probs = out + (size_t)SEQ * DIM;     // second tuple output

  char* w = (char*)d_ws;
  const size_t SB = (size_t)SEQ * DIM * 2;    // bf16 activation
  const size_t SF = (size_t)SEQ * DIM * 4;    // f32 activation
  const size_t SW = (size_t)DIM * DIM * 2;    // bf16 packed weight
  u16* xb   = (u16*)w;  w += SB;
  u16* qb   = (u16*)w;  w += SB;
  u16* kb   = (u16*)w;  w += SB;
  u16* vb   = (u16*)w;  w += SB;
  u16* ab   = (u16*)w;  w += SB;
  u16* hb   = (u16*)w;  w += SB;
  u16* l1b  = (u16*)w;  w += SB;
  u16* Wqb  = (u16*)w;  w += SW;
  u16* Wkb  = (u16*)w;  w += SW;
  u16* Wvb  = (u16*)w;  w += SW;
  u16* Wfcb = (u16*)w;  w += SW;
  u16* W1b  = (u16*)w;  w += SW;
  u16* W2b  = (u16*)w;  w += SW;
  float* fcf  = (float*)w; w += SF;
  float* l1f  = (float*)w; w += SF;
  float* mlpf = (float*)w; w += SF;

  const int NEL = SEQ * DIM;
  pack_bf16<<<(NEL + 255) / 256, 256, 0, stream>>>(x, xb, NEL);
  const int NW = DIM * DIM;
  pack_wT<<<(NW + 255) / 256, 256, 0, stream>>>(Wq,  Wqb);
  pack_wT<<<(NW + 255) / 256, 256, 0, stream>>>(Wk,  Wkb);
  pack_wT<<<(NW + 255) / 256, 256, 0, stream>>>(Wv,  Wvb);
  pack_wT<<<(NW + 255) / 256, 256, 0, stream>>>(Wfc, Wfcb);
  pack_wT<<<(NW + 255) / 256, 256, 0, stream>>>(W1,  W1b);
  pack_wT<<<(NW + 255) / 256, 256, 0, stream>>>(W2,  W2b);

  dim3 ggrid(DIM / 128, SEQ / 128);
  // q scaled by 1/sqrt(DH)=0.125 after bias, per reference
  gemm_bf16<<<ggrid, 256, 0, stream>>>(xb, Wqb, bq, 0.125f, 0, nullptr, qb);
  gemm_bf16<<<ggrid, 256, 0, stream>>>(xb, Wkb, bk, 1.0f,   0, nullptr, kb);
  gemm_bf16<<<ggrid, 256, 0, stream>>>(xb, Wvb, bv, 1.0f,   0, nullptr, vb);

  attn_swa<<<dim3(SEQ / 16, NHEAD), 32, 0, stream>>>(qb, kb, vb, ab, probs);

  gemm_bf16<<<ggrid, 256, 0, stream>>>(ab, Wfcb, bfc, 1.0f, 0, fcf, nullptr);
  ln_fused<<<SEQ, 256, 0, stream>>>(fcf, x, g1, b1, l1f, l1b);
  gemm_bf16<<<ggrid, 256, 0, stream>>>(l1b, W1b, bw1, 1.0f, 1, nullptr, hb);
  gemm_bf16<<<ggrid, 256, 0, stream>>>(hb,  W2b, bw2, 1.0f, 0, mlpf, nullptr);
  ln_fused<<<SEQ, 256, 0, stream>>>(mlpf, l1f, g2, b2, out, nullptr);
}
